// CGC_69836168233304
// MI455X (gfx1250) — matrix operand order repaired and checked by reading, not compiled, not run on hardware
//
#include <hip/hip_runtime.h>

#define Bsz 16384
#define Dd  1024
#define Oo  1024
#define NSTEP (12 * 32)   // 12 experts x 32 K-steps (K-chunk = 32)

typedef __attribute__((ext_vector_type(16))) __bf16 v16bf;
typedef __attribute__((ext_vector_type(8)))  __bf16 v8bf;
typedef __attribute__((ext_vector_type(4)))  __bf16 v4bf;
typedef __attribute__((ext_vector_type(8)))  float  v8f;

// round-to-nearest-even f32 -> bf16 without relying on cvt builtins
static __device__ __forceinline__ __bf16 f2bf(float f) {
    union { float f; unsigned u; } v; v.f = f;
    unsigned r = v.u + 0x7FFFu + ((v.u >> 16) & 1u);
    unsigned short h = (unsigned short)(r >> 16);
    return __builtin_bit_cast(__bf16, h);
}

// gfx1250 async global->LDS copy, 16B per lane, tracked by ASYNCcnt.
// low 32 bits of a flat LDS pointer == LDS byte offset (aperture layout).
static __device__ __forceinline__ void async_copy16(const void* lds_ptr,
                                                    const __bf16* gptr) {
    unsigned lds = (unsigned)(size_t)lds_ptr;
    unsigned long long g = (unsigned long long)(size_t)gptr;
    asm volatile("global_load_async_to_lds_b128 %0, %1, off"
                 :: "v"(lds), "v"(g)
                 : "memory");
}

// ---------------- prep: x f32 -> bf16 (row-major kept) ----------------
__global__ __launch_bounds__(256) void cvt_x_kernel(const float* __restrict__ x,
                                                    __bf16* __restrict__ xb) {
    size_t i = ((size_t)blockIdx.x * 256 + threadIdx.x) * 4;
    float4 f = *(const float4*)(x + i);
    v4bf o; o[0] = f2bf(f.x); o[1] = f2bf(f.y); o[2] = f2bf(f.z); o[3] = f2bf(f.w);
    *(v4bf*)(xb + i) = o;
}

// ---------------- prep: W[m][d][o] f32 -> WT[m][o][d] bf16 (tiled transpose) ---
__global__ __launch_bounds__(256) void wt_kernel(const float* __restrict__ W,
                                                 __bf16* __restrict__ WT) {
    __shared__ __bf16 tile[32][33];
    const int m  = blockIdx.z;
    const int o0 = blockIdx.x * 32, d0 = blockIdx.y * 32;
    const float* src = W + (size_t)m * Dd * Oo;
    for (int r = threadIdx.y; r < 32; r += 8)
        tile[r][threadIdx.x] = f2bf(src[(size_t)(d0 + r) * Oo + o0 + threadIdx.x]);
    __syncthreads();
    __bf16* dst = WT + (size_t)m * Oo * Dd;
    for (int r = threadIdx.y; r < 32; r += 8)
        dst[(size_t)(o0 + r) * Dd + d0 + threadIdx.x] = tile[threadIdx.x][r];
}

// ---------------- prep: gates -> coef[3][B][12] ----------------
__global__ __launch_bounds__(128) void gates_kernel(const float* __restrict__ x,
                                                    const float* __restrict__ Wg,
                                                    const float* __restrict__ bg,
                                                    const float* __restrict__ Wgs,
                                                    const float* __restrict__ bgs,
                                                    float* __restrict__ coef) {
    __shared__ float xs[Dd];
    __shared__ float logit[28];
    const int b = blockIdx.x;
    for (int i = threadIdx.x; i < Dd; i += 128) xs[i] = x[(size_t)b * Dd + i];
    __syncthreads();
    const int j = threadIdx.x;
    if (j < 28) {
        float a;
        if (j < 16) {
            const int t = j >> 3, i = j & 7;
            a = bg[t * 8 + i];
            const float* w = Wg + (size_t)t * Dd * 8 + i;
            for (int d = 0; d < Dd; ++d) a += xs[d] * w[(size_t)d * 8];
        } else {
            const int i = j - 16;
            a = bgs[i];
            for (int d = 0; d < Dd; ++d) a += xs[d] * Wgs[(size_t)d * 12 + i];
        }
        logit[j] = a;
    }
    __syncthreads();
    if (threadIdx.x < 3) {
        const int s = threadIdx.x;
        float vals[12];
        int   mof[12];
        int   n;
        if (s < 2) {
            n = 8;
            for (int i = 0; i < 8; ++i) {
                vals[i] = logit[s * 8 + i];
                mof[i]  = (i < 4) ? (s * 4 + i) : (8 + i - 4);
            }
        } else {
            n = 12;
            for (int i = 0; i < 12; ++i) { vals[i] = logit[16 + i]; mof[i] = i; }
        }
        float mx = vals[0];
        for (int i = 1; i < n; ++i) mx = fmaxf(mx, vals[i]);
        float sum = 0.f;
        for (int i = 0; i < n; ++i) { vals[i] = __expf(vals[i] - mx); sum += vals[i]; }
        const float inv = 1.f / sum;
        float* dst = coef + (size_t)s * Bsz * 12 + (size_t)b * 12;
        for (int m = 0; m < 12; ++m) dst[m] = 0.f;
        for (int i = 0; i < n; ++i) dst[mof[i]] = vals[i] * inv;
    }
}

// ---------------- main fused MoE GEMM ----------------
// 128x128 block tile, 512 threads = 16 wave32. Wave w: rows (w>>1)*16,
// cols (w&1)*64 as 4 tiles of 16. K staged 32-deep into a 4-slot LDS ring
// via async global->LDS copies, prefetch distance 3 (async completes in
// order -> s_wait_asynccnt 4 retires exactly the oldest stage). One
// barrier per K-step. Experts outermost; h never touches memory.
#define APITCH 40  // bf16 per LDS row (64B data + 16B pad -> conflict-free frags)

__global__ __launch_bounds__(512, 1) void moe_kernel(const __bf16* __restrict__ xb,
                                                     const __bf16* __restrict__ WT,
                                                     const float* __restrict__ bias,
                                                     const float* __restrict__ coef,
                                                     float* __restrict__ out) {
    __shared__ __bf16 As[4][128][APITCH];
    __shared__ __bf16 Bs[4][128][APITCH];

    const int tid  = threadIdx.x;
    const int lane = tid & 31;
    const int wv   = tid >> 5;
    const int lm = lane & 15, lh = lane >> 4;
    const int rt = wv >> 1;    // row tile 0..7
    const int ch = wv & 1;     // col half 0..1
    const int b0 = blockIdx.x * 128;
    const int o0 = blockIdx.y * 128;

    v8f acc[3][4];
#pragma unroll
    for (int s = 0; s < 3; ++s)
#pragma unroll
        for (int t = 0; t < 4; ++t) acc[s][t] = (v8f){};

    // issue one K-step's tiles: 512 threads x (1 A + 1 B) 16B async copies
    const int srow = tid >> 2, sc = tid & 3;
    auto stage = [&](int step) {
        const int buf = step & 3;
        const int m2  = step >> 5;
        const int k0  = (step & 31) << 5;
        const __bf16* gA = xb + (size_t)(b0 + srow) * Dd + k0 + sc * 8;
        async_copy16(&As[buf][srow][sc * 8], gA);
        const __bf16* gB = WT + ((size_t)m2 * Oo + o0 + srow) * Dd + k0 + sc * 8;
        async_copy16(&Bs[buf][srow][sc * 8], gB);
    };

    stage(0);
    stage(1);
    stage(2);

    for (int m = 0; m < 12; ++m) {
        v8f h[4];
#pragma unroll
        for (int t = 0; t < 4; ++t) h[t] = (v8f){};

#pragma unroll 4
        for (int kk = 0; kk < 32; ++kk) {
            const int step = m * 32 + kk;
            const int buf  = step & 3;

            // oldest in-flight stage (2 ops) retires when count drops to 4
            asm volatile("s_wait_asynccnt 4" ::: "memory");
            __syncthreads();
            {   // keep issue cadence uniform; wrap-around stages are harmless
                int s2 = step + 3;
                if (s2 >= NSTEP) s2 -= NSTEP;
                stage(s2);
            }

            v16bf a;
            {
                const v8bf alo = *(const v8bf*)&As[buf][rt * 16 + lm][lh * 8];
                const v8bf ahi = *(const v8bf*)&As[buf][rt * 16 + lm][16 + lh * 8];
#pragma unroll
                for (int i = 0; i < 8; ++i) { a[i] = alo[i]; a[8 + i] = ahi[i]; }
            }
#pragma unroll
            for (int t = 0; t < 4; ++t) {
                v16bf bm;
                const int br = ch * 64 + t * 16 + lm;
                const v8bf blo = *(const v8bf*)&Bs[buf][br][lh * 8];
                const v8bf bhi = *(const v8bf*)&Bs[buf][br][16 + lh * 8];
#pragma unroll
                for (int i = 0; i < 8; ++i) { bm[i] = blo[i]; bm[8 + i] = bhi[i]; }
                h[t] = __builtin_amdgcn_wmma_f32_16x16x32_bf16(
                    false, a, false, bm, (short)0, h[t], false, false);
            }
        }

        // per-expert epilogue: bias + relu + gate-weighted accumulate
        float c[3][8];
        const float* cb = coef + (size_t)(b0 + rt * 16 + lh * 8) * 12 + m;
#pragma unroll
        for (int s = 0; s < 3; ++s)
#pragma unroll
            for (int j = 0; j < 8; ++j)
                c[s][j] = cb[(size_t)s * Bsz * 12 + (size_t)j * 12];

#pragma unroll
        for (int t = 0; t < 4; ++t) {
            const float bv = bias[m * Oo + o0 + ch * 64 + t * 16 + lm];
#pragma unroll
            for (int j = 0; j < 8; ++j) {
                float hv = h[t][j] + bv;
                hv = hv > 0.f ? hv : 0.f;
                acc[0][t][j] += c[0][j] * hv;
                acc[1][t][j] += c[1][j] * hv;
                acc[2][t][j] += c[2][j] * hv;
            }
        }
    }

#pragma unroll
    for (int s = 0; s < 3; ++s)
#pragma unroll
        for (int t = 0; t < 4; ++t)
#pragma unroll
            for (int j = 0; j < 8; ++j)
                out[(size_t)s * Bsz * Oo +
                    (size_t)(b0 + rt * 16 + lh * 8 + j) * Oo +
                    o0 + ch * 64 + t * 16 + lm] = acc[s][t][j];
}

extern "C" void kernel_launch(void* const* d_in, const int* in_sizes, int n_in,
                              void* d_out, int out_size, void* d_ws, size_t ws_size,
                              hipStream_t stream) {
    const float* x   = (const float*)d_in[0];
    const float* W   = (const float*)d_in[1];
    const float* bb  = (const float*)d_in[2];
    const float* Wg  = (const float*)d_in[3];
    const float* bg  = (const float*)d_in[4];
    const float* Wgs = (const float*)d_in[5];
    const float* bgs = (const float*)d_in[6];
    float* out = (float*)d_out;

    char* ws = (char*)d_ws;
    __bf16* xb   = (__bf16*)ws;                                   // 32 MB
    __bf16* WT   = (__bf16*)(ws + (size_t)Bsz * Dd * 2);          // 24 MB
    float*  coef = (float*)(ws + (size_t)Bsz * Dd * 2 + (size_t)12 * Oo * Dd * 2);

    cvt_x_kernel<<<dim3((Bsz * Dd) / (256 * 4)), dim3(256), 0, stream>>>(x, xb);
    wt_kernel<<<dim3(Oo / 32, Dd / 32, 12), dim3(32, 8), 0, stream>>>(W, WT);
    gates_kernel<<<dim3(Bsz), dim3(128), 0, stream>>>(x, Wg, bg, Wgs, bgs, coef);
    moe_kernel<<<dim3(Bsz / 128, Oo / 128), dim3(512), 0, stream>>>(xb, WT, bb, coef, out);
}
